// LSTMModel3_43911745634899
// MI455X (gfx1250) — matrix operand order repaired and checked by reading, not compiled, not run on hardware
//
#include <hip/hip_runtime.h>
#include <math.h>

// ---------------------------------------------------------------------------
// LSTM + storage feedback + MLP head, persistent-kernel implementation for
// gfx1250 (MI455X).  bf16 WMMA (v_wmma_f32_16x16x32_bf16) for the two big
// per-step GEMMs; grid barrier between dependent phases instead of 512x3
// kernel launches.
// ---------------------------------------------------------------------------

typedef __attribute__((ext_vector_type(16))) __bf16 v16bf;
typedef __attribute__((ext_vector_type(8)))  float  v8f;

union V8    { v8f   v; float f[8]; };
union FragU { uint4 u[2]; v16bf v; };

constexpr int N   = 256;
constexpr int L   = 512;
constexpr int HIN = 7;
constexpr int H1  = 512;
constexpr int G4  = 4 * H1;          // 2048 gate rows
constexpr int KT  = H1 / 32;         // 16 k-tiles of 32

constexpr int TPB    = 256;          // 8 wave32 per block
constexpr int BLOCKS = 64;           // 512 waves total

// Workspace layout (bytes)
constexpr size_t SZ_WHH  = (size_t)G4 * H1 * 2;     // bf16 fragments of W_hh
constexpr size_t SZ_W1   = (size_t)H1 * H1 * 2;     // bf16 fragments of W1
constexpr size_t SZ_HXA  = (size_t)N  * H1 * 2;     // bf16 A-fragments of hx
constexpr size_t OFF_WHH = 0;
constexpr size_t OFF_W1  = OFF_WHH + SZ_WHH;
constexpr size_t OFF_HXA = OFF_W1  + SZ_W1;         // two buffers
constexpr size_t OFF_CX  = OFF_HXA + 2 * SZ_HXA;
constexpr size_t OFF_H1B = OFF_CX  + (size_t)N * H1 * 4;
constexpr size_t OFF_STOR= OFF_H1B + (size_t)N * H1 * 4;
constexpr size_t OFF_BAR = OFF_STOR + 1024;

__device__ __forceinline__ unsigned pack2bf(float a, float b) {
  unsigned ua = __builtin_bit_cast(unsigned, a);
  unsigned ub = __builtin_bit_cast(unsigned, b);
  ua = (ua + 0x7FFFu + ((ua >> 16) & 1u)) >> 16;           // RNE f32 -> bf16
  ub = (ub + 0x7FFFu + ((ub >> 16) & 1u)) >> 16;
  return (ua & 0xFFFFu) | (ub << 16);
}

__device__ __forceinline__ unsigned short to_bf16(float a) {
  unsigned ua = __builtin_bit_cast(unsigned, a);
  return (unsigned short)((ua + 0x7FFFu + ((ua >> 16) & 1u)) >> 16);
}

__device__ __forceinline__ float sigmoidf_fast(float v) {
  return 1.0f / (1.0f + __expf(-v));
}
__device__ __forceinline__ float tanhf_fast(float v) {
  float a = __expf(-2.0f * fabsf(v));     // in (0,1], numerically stable
  float r = (1.0f - a) / (1.0f + a);
  return copysignf(r, v);
}

// ---------------------------------------------------------------------------
// Prep kernel: build bf16 B-fragments for W_hh and W1.
// B fragment (K=32 x N=16 bf16): dword index = frag*256 + lane*8 + j
//   n = lane & 15 ; k = (lane<16 ? 0 : 8) + (j<4 ? 0 : 16) + 2*(j&3) (+0/+1)
//   frag = n_tile*KT + k_tile
// Also zeroes the grid-barrier state.
// ---------------------------------------------------------------------------
__global__ void prep_frags(const float* __restrict__ Whh,
                           const float* __restrict__ W1,
                           char* __restrict__ ws) {
  const int ND_WHH = G4 * H1 / 2;     // dwords
  const int ND_W1  = H1 * H1 / 2;
  int tid = blockIdx.x * blockDim.x + threadIdx.x;
  if (tid == 0) {
    unsigned* bar = (unsigned*)(ws + OFF_BAR);
    bar[0] = 0u; bar[1] = 0u;
  }
  const float* W; unsigned* out; int idx;
  if (tid < ND_WHH)              { W = Whh; out = (unsigned*)(ws + OFF_WHH); idx = tid; }
  else if (tid < ND_WHH + ND_W1) { W = W1;  out = (unsigned*)(ws + OFF_W1);  idx = tid - ND_WHH; }
  else return;

  int j     = idx & 7;
  int lane  = (idx >> 3) & 31;
  int frag  = idx >> 8;
  int kt    = frag & (KT - 1);
  int ntile = frag / KT;
  int n     = lane & 15;
  int k     = ((lane & 16) ? 8 : 0) + ((j & 4) ? 16 : 0) + 2 * (j & 3);
  const float* src = W + (size_t)(ntile * 16 + n) * H1 + kt * 32 + k;
  out[idx] = pack2bf(src[0], src[1]);
}

// ---------------------------------------------------------------------------
// Grid barrier (sense-reversing, global atomics + s_sleep spin)
// ---------------------------------------------------------------------------
__device__ __forceinline__ void grid_barrier(volatile unsigned* bar) {
  __syncthreads();
  if (threadIdx.x == 0) {
    __threadfence();
    unsigned gen = bar[1];
    unsigned old = atomicAdd((unsigned*)bar, 1u);
    if (old == gridDim.x - 1) {
      bar[0] = 0u;
      __threadfence();
      atomicAdd((unsigned*)(bar + 1), 1u);
    } else {
      while (bar[1] == gen) { __builtin_amdgcn_s_sleep(2); }
    }
    __threadfence();
  }
  __syncthreads();
}

// A-fragment byte offset for hx element (row, h).
// A (M=16 x K=32 bf16): lane<16 -> M=lane, K in {0..7,16..23}; lane>=16 -> K in {8..15,24..31}
__device__ __forceinline__ size_t hxA_byteoff(int row, int h) {
  int kt   = h >> 5;
  int kk   = h & 31;
  int m    = row & 15;
  int lane = ((kk & 15) < 8) ? m : (m + 16);
  int j    = ((kk < 16) ? 0 : 4) + ((kk & 7) >> 1);
  int dw   = ((row >> 4) * KT + kt) * 256 + lane * 8 + j;
  return (size_t)dw * 4 + (size_t)((kk & 1) * 2);
}

// ---------------------------------------------------------------------------
// Persistent kernel: all 512 time steps.
// ---------------------------------------------------------------------------
__global__ void __launch_bounds__(TPB, 1)
lstm_persistent(const float* __restrict__ x,
                const float* __restrict__ W_ih,
                const float* __restrict__ b_ih,
                const float* __restrict__ b_hh,
                const float* __restrict__ b1,
                const float* __restrict__ W2,
                const float* __restrict__ b2,
                float* __restrict__ out,
                char* __restrict__ ws) {
  __shared__ float s_inp[N * 8];     // [row][8] = { x_t[0..6], storage }

  unsigned* hxA0  = (unsigned*)(ws + OFF_HXA);
  unsigned* hxA1  = (unsigned*)(ws + OFF_HXA + SZ_HXA);
  float*    cx    = (float*)(ws + OFF_CX);
  float*    h1buf = (float*)(ws + OFF_H1B);
  float*    stor  = (float*)(ws + OFF_STOR);
  volatile unsigned* bar = (unsigned*)(ws + OFF_BAR);

  float* o_out = out;                   // [N*L] outputs
  float* o_st  = out + (size_t)N * L;   // [N*L] implied storages

  const int lane = threadIdx.x & 31;
  const int wv   = blockIdx.x * (TPB / 32) + (threadIdx.x >> 5);  // 0..511
  const int mt   = wv >> 5;             // batch-row tile 0..15
  const int nt   = wv & 31;             // h / h2 tile 0..31

  // ---- per-call init (harness does not re-zero between replays) ----
  int gtid = blockIdx.x * TPB + threadIdx.x;
  int gsz  = gridDim.x * TPB;
  for (int i = gtid; i < N * H1; i += gsz)     cx[i]   = 0.f;
  for (int i = gtid; i < N * H1 / 2; i += gsz) hxA0[i] = 0u;   // hx(t=-1) = 0
  if (gtid < N) {
    float s = x[(size_t)gtid * L * HIN];       // cur_stor(0) = x(0,0)
    stor[gtid] = s;
    o_st[(size_t)gtid * L] = s;
  }
  grid_barrier(bar);

  const uint4* Bg  = (const uint4*)(ws + OFF_WHH);
  const uint4* Bw1 = (const uint4*)(ws + OFF_W1);

  for (int t = 0; t < L; ++t) {
    // stage [x_t | storage] into LDS
    for (int i = threadIdx.x; i < N * 8; i += TPB) {
      int row = i >> 3, k = i & 7;
      s_inp[i] = (k < HIN) ? x[((size_t)row * L + t) * HIN + k] : stor[row];
    }
    __syncthreads();

    unsigned* hxR = (t & 1) ? hxA1 : hxA0;   // hx(t-1)
    unsigned* hxW = (t & 1) ? hxA0 : hxA1;   // hx(t)

    // ================= Phase B: gates GEMM + LSTM cell =================
    {
      V8 acc[4];
      v8f zero = {};
      #pragma unroll
      for (int q = 0; q < 4; ++q) acc[q].v = zero;

      const uint4* A4 = (const uint4*)hxR;
      #pragma unroll 4
      for (int kt = 0; kt < KT; ++kt) {
        FragU a;
        int ai = (mt * KT + kt) * 64 + lane * 2;
        a.u[0] = A4[ai]; a.u[1] = A4[ai + 1];
        #pragma unroll
        for (int q = 0; q < 4; ++q) {
          FragU b;
          int bi = ((q * 32 + nt) * KT + kt) * 64 + lane * 2;
          b.u[0] = Bg[bi]; b.u[1] = Bg[bi + 1];
          acc[q].v = __builtin_amdgcn_wmma_f32_16x16x32_bf16(
              false, a.v, false, b.v, (short)0, acc[q].v, false, false);
        }
      }

      // bias + input-path contribution (K = HIN+1 = 8, VALU)
      const int nloc = lane & 15;
      const int h    = nt * 16 + nloc;
      const int moff = (lane & 16) ? 8 : 0;
      #pragma unroll
      for (int q = 0; q < 4; ++q) {
        int g = q * H1 + h;
        float bb = b_ih[g] + b_hh[g];
        const float* wr = W_ih + (size_t)g * (HIN + 1);
        float w0 = wr[0], w1 = wr[1], w2 = wr[2], w3 = wr[3];
        float w4 = wr[4], w5 = wr[5], w6 = wr[6], w7 = wr[7];
        #pragma unroll
        for (int r = 0; r < 8; ++r) {
          const float* ip = s_inp + ((mt * 16 + moff + r) << 3);
          acc[q].f[r] += bb + ip[0]*w0 + ip[1]*w1 + ip[2]*w2 + ip[3]*w3
                            + ip[4]*w4 + ip[5]*w5 + ip[6]*w6 + ip[7]*w7;
        }
      }

      // LSTM activations entirely in registers; hx -> bf16 A-fragments
      #pragma unroll
      for (int r = 0; r < 8; ++r) {
        int row = mt * 16 + moff + r;
        float ig = sigmoidf_fast(acc[0].f[r]);
        float fg = sigmoidf_fast(acc[1].f[r]);
        float gg = tanhf_fast  (acc[2].f[r]);
        float og = sigmoidf_fast(acc[3].f[r]);
        size_t ci = (size_t)row * H1 + h;
        float cn = fg * cx[ci] + ig * gg;
        cx[ci] = cn;
        float hv = og * tanhf_fast(cn);
        *(unsigned short*)((char*)hxW + hxA_byteoff(row, h)) = to_bf16(hv);
      }
    }
    grid_barrier(bar);   // hx(t) ready

    // ================= Phase C: h1 = relu(hx @ W1^T + b1) =================
    {
      V8 acc; v8f zero = {}; acc.v = zero;
      const uint4* A4 = (const uint4*)hxW;
      #pragma unroll 4
      for (int kt = 0; kt < KT; ++kt) {
        FragU a, b;
        int ai = (mt * KT + kt) * 64 + lane * 2;
        a.u[0] = A4[ai]; a.u[1] = A4[ai + 1];
        int bi = (nt * KT + kt) * 64 + lane * 2;
        b.u[0] = Bw1[bi]; b.u[1] = Bw1[bi + 1];
        acc.v = __builtin_amdgcn_wmma_f32_16x16x32_bf16(
            false, a.v, false, b.v, (short)0, acc.v, false, false);
      }
      const int nloc = lane & 15;
      const int h2   = nt * 16 + nloc;
      const float bb = b1[h2];
      const int moff = (lane & 16) ? 8 : 0;
      #pragma unroll
      for (int r = 0; r < 8; ++r) {
        float v = acc.f[r] + bb;
        v = v > 0.f ? v : 0.f;
        h1buf[(size_t)(mt * 16 + moff + r) * H1 + h2] = v;
      }
    }
    grid_barrier(bar);   // h1 ready

    // ============ Phase D: out = h1 @ W2^T + b2 ; next storage ============
    if (wv < N) {
      const int row = wv;
      const float* hp = h1buf + (size_t)row * H1;
      float a = 0.f;
      #pragma unroll 4
      for (int k = lane; k < H1; k += 32) a += hp[k] * W2[k];
      #pragma unroll
      for (int off = 16; off; off >>= 1) a += __shfl_xor(a, off, 32);
      if (lane == 0) {
        float ov = a + b2[0];
        o_out[(size_t)row * L + t] = ov;
        if (t + 1 < L) {
          float sn = stor[row] + x[((size_t)row * L + (t + 1)) * HIN] - ov;
          stor[row] = sn;                           // cur_stor(t+1)
          o_st[(size_t)row * L + (t + 1)] = sn;
        }
      }
    }
    grid_barrier(bar);   // storage ready for next step
  }
}

// ---------------------------------------------------------------------------
extern "C" void kernel_launch(void* const* d_in, const int* in_sizes, int n_in,
                              void* d_out, int out_size, void* d_ws, size_t ws_size,
                              hipStream_t stream) {
  const float* x    = (const float*)d_in[0];
  const float* W_ih = (const float*)d_in[1];
  const float* W_hh = (const float*)d_in[2];
  const float* b_ih = (const float*)d_in[3];
  const float* b_hh = (const float*)d_in[4];
  const float* W1   = (const float*)d_in[5];
  const float* b1   = (const float*)d_in[6];
  const float* W2   = (const float*)d_in[7];
  const float* b2   = (const float*)d_in[8];

  const int nd = (G4 * H1 / 2) + (H1 * H1 / 2);
  prep_frags<<<(nd + TPB - 1) / TPB, TPB, 0, stream>>>(W_hh, W1, (char*)d_ws);

  lstm_persistent<<<BLOCKS, TPB, 0, stream>>>(
      x, W_ih, b_ih, b_hh, b1, W2, b2, (float*)d_out, (char*)d_ws);
}
